// TopKSAE_35527969473084
// MI455X (gfx1250) — matrix operand order repaired and checked by reading, not compiled, not run on hardware
//
#include <hip/hip_runtime.h>

#define D_IN   1024
#define D_SAE  16384
#define NROWS  8192
#define TOPK   32
#define KTILE  32

typedef __attribute__((ext_vector_type(16))) __bf16 v16bf;
typedef __attribute__((ext_vector_type(8)))  __bf16 v8bf;
typedef __attribute__((ext_vector_type(4)))  __bf16 v4bf;
typedef __attribute__((ext_vector_type(8)))  float  v8f;

// --- CDNA5 async copy: global -> LDS, 16B per lane, tracked by ASYNCcnt ----
__device__ __forceinline__ void async_b128_to_lds(unsigned lds_addr,
                                                  unsigned long long gaddr) {
    asm volatile("global_load_async_to_lds_b128 %0, %1, off"
                 :: "v"(lds_addr), "v"(gaddr) : "memory");
}
__device__ __forceinline__ void wait_async_0() {
    asm volatile("s_wait_asynccnt 0" ::: "memory");
}

// ---------------------------------------------------------------------------
// Kernel 0: one-shot fp32 -> bf16 convert.  A = bf16(x - b_pre) [N, D_IN],
// B = bf16(W_enc) [D_SAE, D_IN].  Both land in L2-resident scratch.
// ---------------------------------------------------------------------------
__global__ __launch_bounds__(256) void sae_convert_bf16(
    const float* __restrict__ x, const float* __restrict__ b_pre,
    const float* __restrict__ W_enc, __bf16* __restrict__ Abf,
    __bf16* __restrict__ Bbf)
{
    const size_t ACH = (size_t)NROWS * D_IN / 4;          // float4 chunks of A
    const size_t TCH = ACH + (size_t)D_SAE * D_IN / 4;    // + chunks of B
    const size_t stride = (size_t)gridDim.x * 256;
    for (size_t c = (size_t)blockIdx.x * 256 + threadIdx.x; c < TCH; c += stride) {
        if (c < ACH) {
            const size_t e = c * 4;
            const int k = (int)(e & (D_IN - 1));
            const float4 xv = *(const float4*)(x + e);
            const float4 bp = *(const float4*)(b_pre + k);
            v4bf o = {(__bf16)(xv.x - bp.x), (__bf16)(xv.y - bp.y),
                      (__bf16)(xv.z - bp.z), (__bf16)(xv.w - bp.w)};
            *(v4bf*)(Abf + e) = o;
        } else {
            const size_t e = (c - ACH) * 4;
            const float4 wv = *(const float4*)(W_enc + e);
            v4bf o = {(__bf16)wv.x, (__bf16)wv.y, (__bf16)wv.z, (__bf16)wv.w};
            *(v4bf*)(Bbf + e) = o;
        }
    }
}

// ---------------------------------------------------------------------------
// Kernel 1: z_pre[N, D_SAE] = A @ B^T  (bf16 WMMA, fp32 accum).
// Block: 256 threads = 8 waves. Macro tile 128x128. Wave strip 16x128.
// Double-buffered LDS staging via GLOBAL_LOAD_ASYNC_TO_LDS_B128 (ASYNCcnt).
// Fragment loads batched ahead of the WMMA burst so DScnt drains with
// counted (partial) waits instead of one full stall per tile.
// ---------------------------------------------------------------------------
__global__ __launch_bounds__(256) void sae_encode_wmma(
    const __bf16* __restrict__ Abf, const __bf16* __restrict__ Bbf,
    float* __restrict__ z_pre)
{
    __shared__ __align__(16) __bf16 Als[2][128][40]; // padded rows: 80B stride
    __shared__ __align__(16) __bf16 Bls[2][128][40];
    const unsigned BUFSTRIDE = 128u * 40u * 2u;      // 10240 B per buffer

    const int tid  = threadIdx.x;
    const int wave = tid >> 5;
    const int lane = tid & 31;
    const int m    = lane & 15;
    const int g    = lane >> 4;

    const int rowBase = blockIdx.x * 128;
    const int colBase = blockIdx.y * 128;

    v8f acc[8];
#pragma unroll
    for (int t = 0; t < 8; ++t) acc[t] = (v8f){0.f,0.f,0.f,0.f,0.f,0.f,0.f,0.f};

    // Async-copy lane mapping: each B128 op moves 8 rows x 64B (32 lanes x 16B).
    // Wave w stages rows [w*16, w*16+16) of both tiles -> 2+2 ops per K-step.
    const int crow  = lane >> 2;           // row within 8-row slab
    const int cbyte = (lane & 3) << 4;     // 16B column chunk
    const int r0    = (wave << 4) + crow;  // slabs r0 and r0+8

    unsigned long long gA0 = (unsigned long long)(uintptr_t)Abf +
                             ((unsigned long long)(rowBase + r0) << 11) + cbyte;
    unsigned long long gA1 = gA0 + (8ull << 11);
    unsigned long long gB0 = (unsigned long long)(uintptr_t)Bbf +
                             ((unsigned long long)(colBase + r0) << 11) + cbyte;
    unsigned long long gB1 = gB0 + (8ull << 11);

    const unsigned baseA = (unsigned)(uintptr_t)&Als[0][0][0];
    const unsigned baseB = (unsigned)(uintptr_t)&Bls[0][0][0];
    const unsigned lA0 = baseA + (unsigned)r0 * 80u + (unsigned)cbyte;
    const unsigned lA1 = lA0 + 8u * 80u;
    const unsigned lB0 = baseB + (unsigned)r0 * 80u + (unsigned)cbyte;
    const unsigned lB1 = lB0 + 8u * 80u;

    // Prologue: stage K-tile 0 into buffer 0
    async_b128_to_lds(lA0, gA0);
    async_b128_to_lds(lA1, gA1);
    async_b128_to_lds(lB0, gB0);
    async_b128_to_lds(lB1, gB1);
    gA0 += 64; gA1 += 64; gB0 += 64; gB1 += 64;
    wait_async_0();
    __syncthreads();

    int p = 0;
    for (int k0 = 0; k0 < D_IN; k0 += KTILE) {
        // Kick off next tile into the other buffer (overlaps with compute)
        if (k0 + KTILE < D_IN) {
            const unsigned q = (unsigned)(p ^ 1) * BUFSTRIDE;
            async_b128_to_lds(lA0 + q, gA0);
            async_b128_to_lds(lA1 + q, gA1);
            async_b128_to_lds(lB0 + q, gB0);
            async_b128_to_lds(lB1 + q, gB1);
            gA0 += 64; gA1 += 64; gB0 += 64; gB1 += 64;
        }

        // A fragment (16x32 bf16): halves 0..7 = K[g*8..+7], 8..15 = K[16+g*8..]
        const int am = (wave << 4) + m;
        const v8bf alo = *(const v8bf*)(&Als[p][am][g * 8]);
        const v8bf ahi = *(const v8bf*)(&Als[p][am][16 + g * 8]);
        v16bf a;
#pragma unroll
        for (int h = 0; h < 8; ++h) { a[h] = alo[h]; a[h + 8] = ahi[h]; }

        // Batch all 8 B fragments first (16 outstanding ds_load_b128),
        // then burst the WMMAs -> counted DScnt waits, load/WMMA overlap.
        v16bf bfr[8];
#pragma unroll
        for (int t = 0; t < 8; ++t) {
            const int n = (t << 4) + m;   // column n = t*16+m, halves K[g*8..+7], K[16+g*8..+7]
            const v8bf blo = *(const v8bf*)(&Bls[p][n][g * 8]);
            const v8bf bhi = *(const v8bf*)(&Bls[p][n][16 + g * 8]);
#pragma unroll
            for (int h = 0; h < 8; ++h) { bfr[t][h] = blo[h]; bfr[t][h + 8] = bhi[h]; }
        }
#pragma unroll
        for (int t = 0; t < 8; ++t) {
            acc[t] = __builtin_amdgcn_wmma_f32_16x16x32_bf16(
                false, a, false, bfr[t], (short)0, acc[t], false, false);
        }

        wait_async_0();   // our async writes into buf p^1 are complete
        __syncthreads();  // waits own DScnt then barrier: all reads of buf p done
        p ^= 1;
    }

    // Store D tiles: lane holds N = m, M = r + 8*g for VGPR r.
    const int outRow0 = rowBase + (wave << 4) + (g << 3);
#pragma unroll
    for (int t = 0; t < 8; ++t) {
        const int col = colBase + (t << 4) + m;
#pragma unroll
        for (int r = 0; r < 8; ++r) {
            z_pre[(size_t)(outRow0 + r) * D_SAE + col] = acc[t][r];
        }
    }
}

// ---------------------------------------------------------------------------
// Kernel 2: per-row top-32 of z_pre; writes dense z and compact (idx,val).
// ---------------------------------------------------------------------------
__global__ __launch_bounds__(256) void sae_topk(
    const float* __restrict__ z_pre, float* __restrict__ z,
    int* __restrict__ ws_idx, float* __restrict__ ws_val)
{
    const int row = blockIdx.x;
    const int tid = threadIdx.x;
    const float* zr = z_pre + (size_t)row * D_SAE;

    float v[64];
#pragma unroll
    for (int e = 0; e < 64; ++e) v[e] = zr[e * 256 + tid];

    unsigned long long claimed = 0ull;
    float lmax = -3.4e38f;
    int   lidx = 0;
#pragma unroll
    for (int e = 0; e < 64; ++e) {
        if (v[e] > lmax) { lmax = v[e]; lidx = e; }
    }

    __shared__ float wmax[8];
    __shared__ int   wtid[8];
    __shared__ int   btidS;

    for (int it = 0; it < TOPK; ++it) {
        float mv = lmax;
        int   mt = tid;
#pragma unroll
        for (int s = 16; s > 0; s >>= 1) {
            const float ov = __shfl_xor(mv, s, 32);
            const int   ot = __shfl_xor(mt, s, 32);
            if (ov > mv) { mv = ov; mt = ot; }
        }
        if ((tid & 31) == 0) { wmax[tid >> 5] = mv; wtid[tid >> 5] = mt; }
        __syncthreads();
        if (tid == 0) {
            float bm = wmax[0]; int bt = wtid[0];
#pragma unroll
            for (int w = 1; w < 8; ++w)
                if (wmax[w] > bm) { bm = wmax[w]; bt = wtid[w]; }
            btidS = bt;
        }
        __syncthreads();
        if (tid == btidS) {
            ws_idx[row * TOPK + it] = lidx * 256 + tid;
            ws_val[row * TOPK + it] = lmax;
            claimed |= (1ull << lidx);
            lmax = -3.4e38f; lidx = 0;
#pragma unroll
            for (int e = 0; e < 64; ++e) {
                const bool taken = (claimed >> e) & 1ull;
                if (!taken && v[e] > lmax) { lmax = v[e]; lidx = e; }
            }
        }
        __syncthreads();
    }

    float* zo = z + (size_t)row * D_SAE;
#pragma unroll
    for (int e = 0; e < 64; ++e) {
        zo[e * 256 + tid] = ((claimed >> e) & 1ull) ? v[e] : 0.0f;
    }
}

// ---------------------------------------------------------------------------
// Kernel 3: sparse decode x_hat = b_dec + b_pre + sum_s val_s * W_dec[:, idx_s]
// ---------------------------------------------------------------------------
__global__ __launch_bounds__(256) void sae_decode_sparse(
    const float* __restrict__ W_dec, const float* __restrict__ b_pre,
    const float* __restrict__ b_dec, const int* __restrict__ ws_idx,
    const float* __restrict__ ws_val, float* __restrict__ x_hat)
{
    const int row = blockIdx.x;
    const int tid = threadIdx.x;

    __shared__ int   sidx[TOPK];
    __shared__ float sval[TOPK];
    if (tid < TOPK) {
        sidx[tid] = ws_idx[row * TOPK + tid];
        sval[tid] = ws_val[row * TOPK + tid];
    }
    __syncthreads();

    float acc[4];
#pragma unroll
    for (int p = 0; p < 4; ++p) {
        const int d = tid + p * 256;
        acc[p] = b_dec[d] + b_pre[d];
    }
    for (int s = 0; s < TOPK; ++s) {
        const int   j  = sidx[s];
        const float vv = sval[s];
#pragma unroll
        for (int p = 0; p < 4; ++p) {
            const int d = tid + p * 256;
            acc[p] = fmaf(vv, W_dec[(size_t)d * D_SAE + j], acc[p]);
        }
    }
#pragma unroll
    for (int p = 0; p < 4; ++p)
        x_hat[(size_t)row * D_IN + tid + p * 256] = acc[p];
}

// ---------------------------------------------------------------------------
extern "C" void kernel_launch(void* const* d_in, const int* in_sizes, int n_in,
                              void* d_out, int out_size, void* d_ws, size_t ws_size,
                              hipStream_t stream) {
    const float* x     = (const float*)d_in[0];
    const float* b_pre = (const float*)d_in[1];
    const float* W_enc = (const float*)d_in[2];
    const float* W_dec = (const float*)d_in[3];
    const float* b_dec = (const float*)d_in[4];

    float* out   = (float*)d_out;
    float* x_hat = out;                                   // [N, D_IN]
    float* z     = out + (size_t)NROWS * D_IN;            // [N, D_SAE]
    float* z_pre = z   + (size_t)NROWS * D_SAE;           // [N, D_SAE]

    // bf16 GEMM operands live in the z output region (512MB) as transient
    // scratch: written by K0, consumed by K1, overwritten by K2's final z.
    __bf16* Abf = (__bf16*)z;                             // 16 MB
    __bf16* Bbf = (__bf16*)z + (size_t)NROWS * D_IN;      // 32 MB

    int*   ws_idx = (int*)d_ws;                                        // [N, 32]
    float* ws_val = (float*)((char*)d_ws + (size_t)NROWS * TOPK * 4);  // [N, 32]

    sae_convert_bf16<<<6144, 256, 0, stream>>>(x, b_pre, W_enc, Abf, Bbf);
    dim3 encGrid(NROWS / 128, D_SAE / 128, 1);
    sae_encode_wmma<<<encGrid, 256, 0, stream>>>(Abf, Bbf, z_pre);
    sae_topk<<<NROWS, 256, 0, stream>>>(z_pre, z, ws_idx, ws_val);
    sae_decode_sparse<<<NROWS, 256, 0, stream>>>(W_dec, b_pre, b_dec, ws_idx, ws_val, x_hat);
}
